// Qwen3_5Attention_2980707303638
// MI455X (gfx1250) — matrix operand order repaired and checked by reading, not compiled, not run on hardware
//
#include <hip/hip_runtime.h>
#include <hip/hip_bf16.h>

// ---------------------------------------------------------------------------
// Qwen3.5 gated attention block for gfx1250 (MI455X).
// bf16 WMMA pipeline, double-buffered TDM LDS staging, K-major B-operand
// layouts; A fragments use wide b128 loads, B fragments scalar gathers.
// B=2, S=2048, D=2048, H=16, KVH=4, HD=128
// ---------------------------------------------------------------------------

typedef __attribute__((ext_vector_type(16))) __bf16 v16bf;
typedef __attribute__((ext_vector_type(8)))  __bf16 v8bf;
typedef __attribute__((ext_vector_type(8)))  float  v8f;
typedef __attribute__((ext_vector_type(4)))  unsigned v4u;
typedef __attribute__((ext_vector_type(8)))  int      v8i;
typedef __attribute__((ext_vector_type(4)))  int      v4i;

#if defined(__has_builtin)
#if __has_builtin(__builtin_amdgcn_tensor_load_to_lds) && \
    __has_builtin(__builtin_amdgcn_s_wait_tensorcnt)
#define HAVE_TDM 1
#endif
#endif
#ifndef HAVE_TDM
#define HAVE_TDM 0
#endif

static __device__ __forceinline__ __bf16 f2bf(float f) {
  unsigned u = __float_as_uint(f);
  unsigned r = (u + 0x7FFFu + ((u >> 16) & 1u)) >> 16;
  unsigned short s = (unsigned short)r;
  __bf16 b;
  __builtin_memcpy(&b, &s, 2);
  return b;
}

static __device__ __forceinline__ float bf2f(__bf16 b) {
  unsigned short s;
  __builtin_memcpy(&s, &b, 2);
  return __uint_as_float(((unsigned)s) << 16);
}

static __device__ __forceinline__ v16bf cat8(v8bf lo, v8bf hi) {
  return __builtin_shufflevector(lo, hi, 0, 1, 2, 3, 4, 5, 6, 7,
                                 8, 9, 10, 11, 12, 13, 14, 15);
}

#if HAVE_TDM
// Build a 2-D TDM descriptor (D#) and issue TENSOR_LOAD_TO_LDS.
// Copies `rows` rows of `row_elems` bf16 elements (source row stride
// `row_stride_elems`) from `gaddr` into LDS at byte offset `lds_off`,
// packed contiguously. 6-arg builtin form (clang-23 / therock-10.0).
static __device__ __forceinline__ void tdm_load_2d(const void* gaddr,
                                                   unsigned lds_off,
                                                   unsigned rows,
                                                   unsigned row_elems,
                                                   unsigned row_stride_elems) {
  unsigned long long ga = (unsigned long long)(size_t)gaddr;
  v4u g0;
  g0[0] = 1u;                                   // count=1, user descriptor
  g0[1] = lds_off;                              // lds_addr (bytes)
  g0[2] = (unsigned)ga;                         // global_addr[31:0]
  g0[3] = (unsigned)((ga >> 32) & 0x1FFFFFFu)   // global_addr[56:32]
          | (2u << 30);                         // type=2 ("image")
  unsigned td0 = row_stride_elems;              // tensor_dim0 (>= tile_dim0)
  unsigned td1 = rows;                          // tensor_dim1
  v8i g1;
  g1[0] = (int)(1u << 16);                      // wg_mask=0, data_size=1 (2B)
  g1[1] = (int)((td0 & 0xFFFFu) << 16);         // tensor_dim0[15:0]
  g1[2] = (int)(((td0 >> 16) & 0xFFFFu) | ((td1 & 0xFFFFu) << 16));
  g1[3] = (int)(((td1 >> 16) & 0xFFFFu) | ((row_elems & 0xFFFFu) << 16)); // tile_dim0
  g1[4] = (int)(rows & 0xFFFFu);                // tile_dim1 (tile_dim2=0)
  g1[5] = (int)row_stride_elems;                // tensor_dim0_stride[31:0]
  g1[6] = 0;
  g1[7] = 0;
  __builtin_amdgcn_tensor_load_to_lds(g0, g1, (v4i)0, (v4i)0, (v8i)0, 0);
}
#endif

// ---------------------------------------------------------------------------
// fp32 -> bf16 conversion (plain, and transposing for K-major weights)
// ---------------------------------------------------------------------------
__global__ __launch_bounds__(256) void cvt_f32_bf16(const float* __restrict__ in,
                                                    __bf16* __restrict__ out, int n) {
  int i = blockIdx.x * 256 + threadIdx.x;
  if (i < n) out[i] = f2bf(in[i]);
}

// in: [Nr][Kc] row-major fp32  ->  out: [Kc][Nr] bf16 (K-major)
__global__ __launch_bounds__(256) void cvt_f32_bf16_T(const float* __restrict__ in,
                                                      __bf16* __restrict__ out,
                                                      int Nr, int Kc) {
  int i = blockIdx.x * 256 + threadIdx.x;
  if (i < Nr * Kc) {
    int n = i / Kc, kk = i % Kc;
    out[(size_t)kk * Nr + n] = f2bf(in[i]);
  }
}

// ---------------------------------------------------------------------------
// GEMM: C[M,N] = A[M,K] @ Wt[K,N] ; A row-major, Wt K-major, both bf16.
// Block = 128 threads (4 waves), C tile 64(M) x 128(N), K-step 32,
// K-loop unrolled 2x across the ping-pong LDS buffers (K % 64 == 0).
// Double-buffered staging via the TDM: wave0 issues the next tile's two
// tensor_load_to_lds ops, then s_wait_tensorcnt(2) -- per-wave tensor ops
// complete in order, so <=2 outstanding means the current buffer is ready --
// and the WMMAs overlap the DMA of the next buffer.
// A-frag: lane=M, elems 0-7 = K[kh*8..], elems 8-15 = K[16+kh*8..] (2 b128)
// B-frag: lane=N, elems 0-7 = K[kh*8..], elems 8-15 = K[16+kh*8..] (scalar loads)
// ---------------------------------------------------------------------------
template <bool F32OUT>
__global__ __launch_bounds__(128) void gemm_bf16_wmma(const __bf16* __restrict__ A,
                                                      const __bf16* __restrict__ Wt,
                                                      void* __restrict__ out,
                                                      int M, int N, int K) {
  __shared__ __bf16 lA[2][64 * 32];    // 2 x 4 KB
  __shared__ __bf16 lB[2][32 * 128];   // 2 x 8 KB

  const int tid   = threadIdx.x;
  const int wave  = tid >> 5;
  const int lane  = tid & 31;
  const int mrow  = lane & 15;
  const int khalf = lane >> 4;
  const int m0    = blockIdx.y * 64;
  const int n0    = blockIdx.x * 128;

  v8f acc[4][2] = {};

  auto compute = [&](int p) {
    v16bf af[4];
#pragma unroll
    for (int mt = 0; mt < 4; ++mt) {
      const __bf16* ap = &lA[p][(mt * 16 + mrow) * 32 + khalf * 8];
      af[mt] = cat8(*(const v8bf*)ap, *(const v8bf*)(ap + 16));
    }
#pragma unroll
    for (int nt = 0; nt < 2; ++nt) {
      v16bf bfrag;
#pragma unroll
      for (int s = 0; s < 16; ++s)
        bfrag[s] = lB[p][(khalf * 8 + s + (s & 8)) * 128 + wave * 32 + nt * 16 + mrow];
#pragma unroll
      for (int mt = 0; mt < 4; ++mt) {
        acc[mt][nt] = __builtin_amdgcn_wmma_f32_16x16x32_bf16(
            false, af[mt], false, bfrag, (short)0, acc[mt][nt], false, false);
      }
    }
  };

#if HAVE_TDM
  auto issue = [&](int p, int k) {
    tdm_load_2d(A + (size_t)m0 * K + k, (unsigned)(size_t)&lA[p][0], 64, 32, K);
    tdm_load_2d(Wt + (size_t)k * N + n0, (unsigned)(size_t)&lB[p][0], 32, 128, N);
  };
  if (wave == 0) issue(0, 0);
#else
  auto stage = [&](int p, int k) {
    if (tid < 64) {
      const __bf16* src = A + (size_t)(m0 + tid) * K + k;
#pragma unroll
      for (int c = 0; c < 4; ++c)
        *(v8bf*)&lA[p][tid * 32 + c * 8] = *(const v8bf*)(src + c * 8);
    }
    int r = tid >> 2, c = (tid & 3) * 32;
    const __bf16* src = Wt + (size_t)(k + r) * N + n0 + c;
#pragma unroll
    for (int q8 = 0; q8 < 4; ++q8)
      *(v8bf*)&lB[p][r * 128 + c + q8 * 8] = *(const v8bf*)(src + q8 * 8);
  };
#endif

  for (int k = 0; k < K; k += 64) {
    // ---- phase 0: buffer 0 holds [k, k+32) ----
#if HAVE_TDM
    if (wave == 0) {
      issue(1, k + 32);                      // prefetch next into buffer 1
      __builtin_amdgcn_s_wait_tensorcnt(2);  // buffer 0 pair has landed
    }
    __syncthreads();
#else
    stage(0, k);
    __syncthreads();
#endif
    compute(0);
    __syncthreads();

    // ---- phase 1: buffer 1 holds [k+32, k+64) ----
#if HAVE_TDM
    if (wave == 0) {
      if (k + 64 < K) {
        issue(0, k + 64);
        __builtin_amdgcn_s_wait_tensorcnt(2);
      } else {
        __builtin_amdgcn_s_wait_tensorcnt(0);
      }
    }
    __syncthreads();
#else
    stage(1, k + 32);
    __syncthreads();
#endif
    compute(1);
    __syncthreads();
  }

  // C frag: elem r -> M = r + 8*khalf ; N = mrow
#pragma unroll
  for (int mt = 0; mt < 4; ++mt) {
#pragma unroll
    for (int nt = 0; nt < 2; ++nt) {
#pragma unroll
      for (int r = 0; r < 8; ++r) {
        size_t m = (size_t)(m0 + mt * 16 + r + 8 * khalf);
        size_t n = (size_t)(n0 + wave * 32 + nt * 16 + mrow);
        if (F32OUT)
          ((float*)out)[m * N + n] = acc[mt][nt][r];
        else
          ((__bf16*)out)[m * N + n] = f2bf(acc[mt][nt][r]);
      }
    }
  }
}

// ---------------------------------------------------------------------------
// Per-head RMSNorm + RoPE. One wave per (b,s,head) row of 128 elements.
// transpose_out=0: out[row*128 + d]                (q path)
// transpose_out=1: out[((b*NH+h)*128 + d)*S + s]   (kT, d-major for B-frags)
// ---------------------------------------------------------------------------
__global__ __launch_bounds__(128) void normrope(const __bf16* __restrict__ in,
                                                __bf16* __restrict__ outp,
                                                const float* __restrict__ normw,
                                                const float* __restrict__ cosp,
                                                const float* __restrict__ sinp,
                                                int nheads, int in_stride,
                                                int total_rows, int transpose_out,
                                                int S) {
  int row = blockIdx.x * 4 + (threadIdx.x >> 5);
  if (row >= total_rows) return;
  int lane = threadIdx.x & 31;
  int bs = row / nheads;  // b*S + s
  int hh = row % nheads;

  const __bf16* rp = in + (size_t)row * in_stride;
  float x[4];
  float ss = 0.f;
#pragma unroll
  for (int t = 0; t < 4; ++t) {
    x[t] = bf2f(rp[lane * 4 + t]);
    ss += x[t] * x[t];
  }
#pragma unroll
  for (int off = 16; off >= 1; off >>= 1) ss += __shfl_xor(ss, off, 32);
  float rms = rsqrtf(ss * (1.0f / 128.0f) + 1e-6f);

  float xn[4];
#pragma unroll
  for (int t = 0; t < 4; ++t) {
    int d = lane * 4 + t;
    xn[t] = x[t] * rms * (1.0f + normw[d]);
  }

  float sgn = (lane < 16) ? -1.0f : 1.0f;  // rotate_half: [-x2, x1]
#pragma unroll
  for (int t = 0; t < 4; ++t) {
    int d = lane * 4 + t;
    float partner = __shfl_xor(xn[t], 16, 32);
    float c = cosp[(size_t)bs * 128 + d];
    float s = sinp[(size_t)bs * 128 + d];
    float val = xn[t] * c + sgn * partner * s;
    if (!transpose_out) {
      outp[(size_t)row * 128 + d] = f2bf(val);
    } else {
      int b = bs / S, spos = bs % S;
      outp[((size_t)(b * nheads + hh) * 128 + d) * S + spos] = f2bf(val);
    }
  }
}

// ---------------------------------------------------------------------------
// Causal GQA flash attention + sigmoid gate.
// grid = (S/16, H, B), one wave per 16 query rows.
// q  : [B,S,H,HD] row-major bf16
// kT : [B,KVH,HD,S] d-major bf16  (B-frag = 16 scalar loads, stride S)
// v  : [B,S,KVH,HD] row-major bf16
// All fragments for a 32-k block are preloaded into registers before the
// WMMA bursts so the loads issue as clauses and overlap the matrix pipe.
// ---------------------------------------------------------------------------
__global__ __launch_bounds__(32) void flash_attn(const __bf16* __restrict__ q,
                                                 const __bf16* __restrict__ kT,
                                                 const __bf16* __restrict__ v,
                                                 const __bf16* __restrict__ qg,
                                                 __bf16* __restrict__ attn) {
  constexpr int S = 2048, H = 16, KVH = 4, HD = 128;
  const int lane  = threadIdx.x;
  const int mrow  = lane & 15;
  const int khalf = lane >> 4;
  const int m0  = blockIdx.x * 16;
  const int h   = blockIdx.y;
  const int b   = blockIdx.z;
  const int kvh = h / (H / KVH);

  __shared__ __bf16 pl[16 * 32];  // P tile restage: C-frag -> A-frag layout

  v16bf qa[4];
#pragma unroll
  for (int kk = 0; kk < 4; ++kk) {
    const __bf16* ap =
        q + ((size_t)(b * S + m0 + mrow) * H + h) * HD + kk * 32 + khalf * 8;
    qa[kk] = cat8(*(const v8bf*)ap, *(const v8bf*)(ap + 16));
  }

  v8f o[8] = {};
  float mrun[8], lrun[8];
#pragma unroll
  for (int r = 0; r < 8; ++r) { mrun[r] = -1e30f; lrun[r] = 0.f; }

  const float sc = 0.08838834764831845f;  // HD^-0.5
  const __bf16* ktbase = kT + (size_t)(b * KVH + kvh) * HD * S;

  for (int kb = 0; kb <= m0 + 15; kb += 32) {
    // Prefetch next K/V block into GL2 while this block computes.
    if (kb + 32 <= m0 + 15) {
      __builtin_prefetch(ktbase + (size_t)lane * S + kb + 32, 0, 1);
      __builtin_prefetch(
          v + ((size_t)(b * S + kb + 32 + lane) * KVH + kvh) * HD, 0, 1);
    }

    // ---- preload all 8 score B-frags, then the 8 WMMAs ----
    v16bf kf[2][4];
#pragma unroll
    for (int h2 = 0; h2 < 2; ++h2) {
#pragma unroll
      for (int kk = 0; kk < 4; ++kk) {
#pragma unroll
        for (int s = 0; s < 16; ++s)
          kf[h2][kk][s] = ktbase[(size_t)(kk * 32 + khalf * 8 + s + (s & 8)) * S + kb + h2 * 16 + mrow];
      }
    }
    v8f s01[2];
#pragma unroll
    for (int h2 = 0; h2 < 2; ++h2) {
      v8f c = {};
#pragma unroll
      for (int kk = 0; kk < 4; ++kk) {
        c = __builtin_amdgcn_wmma_f32_16x16x32_bf16(
            false, qa[kk], false, kf[h2][kk], (short)0, c, false, false);
      }
      s01[h2] = c;
    }

#pragma unroll
    for (int r = 0; r < 8; ++r) {
      int qa_row = m0 + r + 8 * khalf;
      s01[0][r] = (kb + mrow      <= qa_row) ? s01[0][r] * sc : -1e30f;
      s01[1][r] = (kb + 16 + mrow <= qa_row) ? s01[1][r] * sc : -1e30f;
    }

    float corr[8];
#pragma unroll
    for (int r = 0; r < 8; ++r) {
      float mx = fmaxf(s01[0][r], s01[1][r]);
#pragma unroll
      for (int off = 8; off >= 1; off >>= 1) mx = fmaxf(mx, __shfl_xor(mx, off, 32));
      float nm = fmaxf(mrun[r], mx);
      corr[r] = __expf(mrun[r] - nm);
      mrun[r] = nm;
      float p0 = __expf(s01[0][r] - nm);
      float p1 = __expf(s01[1][r] - nm);
      s01[0][r] = p0;
      s01[1][r] = p1;
      float rs = p0 + p1;
#pragma unroll
      for (int off = 8; off >= 1; off >>= 1) rs += __shfl_xor(rs, off, 32);
      lrun[r] = lrun[r] * corr[r] + rs;
    }
#pragma unroll
    for (int n = 0; n < 8; ++n)
#pragma unroll
      for (int r = 0; r < 8; ++r) o[n][r] *= corr[r];

    __syncthreads();
#pragma unroll
    for (int r = 0; r < 8; ++r) {
      int row = r + 8 * khalf;
      pl[row * 32 + mrow]      = f2bf(s01[0][r]);
      pl[row * 32 + 16 + mrow] = f2bf(s01[1][r]);
    }
    __syncthreads();
    const __bf16* pp = &pl[mrow * 32 + khalf * 8];
    v16bf pa = cat8(*(const v8bf*)pp, *(const v8bf*)(pp + 16));

    // ---- preload all 8 V-frags, then the 8 WMMAs ----
    v16bf vb[8];
#pragma unroll
    for (int n = 0; n < 8; ++n) {
#pragma unroll
      for (int s = 0; s < 16; ++s)
        vb[n][s] = v[((size_t)(b * S + kb + khalf * 8 + s + (s & 8)) * KVH + kvh) * HD + n * 16 + mrow];
    }
#pragma unroll
    for (int n = 0; n < 8; ++n) {
      o[n] = __builtin_amdgcn_wmma_f32_16x16x32_bf16(false, pa, false, vb[n],
                                                     (short)0, o[n], false, false);
    }
  }

#pragma unroll
  for (int r = 0; r < 8; ++r) {
    int q_abs = m0 + r + 8 * khalf;
    float inv = 1.0f / lrun[r];
#pragma unroll
    for (int n = 0; n < 8; ++n) {
      int d = n * 16 + mrow;
      float val = o[n][r] * inv;
      float g = bf2f(qg[((size_t)(b * S + q_abs) * H + h) * 256 + 128 + d]);
      val *= 1.0f / (1.0f + __expf(-g));
      attn[(size_t)(b * S + q_abs) * 2048 + h * 128 + d] = f2bf(val);
    }
  }
}

// ---------------------------------------------------------------------------
// Host-side orchestration
// ---------------------------------------------------------------------------
extern "C" void kernel_launch(void* const* d_in, const int* in_sizes, int n_in,
                              void* d_out, int out_size, void* d_ws, size_t ws_size,
                              hipStream_t stream) {
  constexpr int B = 2, S = 2048, D = 2048, H = 16, KVH = 4, HD = 128;
  constexpr int M = B * S;          // 4096
  constexpr int NQ = H * HD * 2;    // 4096
  constexpr int NKV = KVH * HD;     // 512

  const float* hs  = (const float*)d_in[0];
  const float* cs  = (const float*)d_in[1];
  const float* sn  = (const float*)d_in[2];
  const float* qw  = (const float*)d_in[3];
  const float* kw  = (const float*)d_in[4];
  const float* vw  = (const float*)d_in[5];
  const float* ow  = (const float*)d_in[6];
  const float* qnw = (const float*)d_in[7];
  const float* knw = (const float*)d_in[8];

  char* wsb = (char*)d_ws;
  size_t off = 0;
  auto take = [&](size_t bytes) { char* p = wsb + off; off += bytes; return p; };
  __bf16* hs_bf = (__bf16*)take((size_t)M * D * 2);         // A (row-major)
  __bf16* qwT   = (__bf16*)take((size_t)D * NQ * 2);        // [D][NQ] K-major
  __bf16* kwT   = (__bf16*)take((size_t)D * NKV * 2);       // [D][NKV]
  __bf16* vwT   = (__bf16*)take((size_t)D * NKV * 2);       // [D][NKV]
  __bf16* owT   = (__bf16*)take((size_t)(H * HD) * D * 2);  // [HHD][D]
  __bf16* qg_bf = (__bf16*)take((size_t)M * NQ * 2);
  __bf16* kr_bf = (__bf16*)take((size_t)M * NKV * 2);
  __bf16* q_bf  = (__bf16*)take((size_t)M * H * HD * 2);
  __bf16* kT_bf = (__bf16*)take((size_t)B * KVH * HD * S * 2);
  __bf16* v_bf  = (__bf16*)take((size_t)M * NKV * 2);
  __bf16* at_bf = (__bf16*)take((size_t)M * H * HD * 2);

  cvt_f32_bf16<<<(M * D + 255) / 256, 256, 0, stream>>>(hs, hs_bf, M * D);
  cvt_f32_bf16_T<<<(NQ * D + 255) / 256, 256, 0, stream>>>(qw, qwT, NQ, D);
  cvt_f32_bf16_T<<<(NKV * D + 255) / 256, 256, 0, stream>>>(kw, kwT, NKV, D);
  cvt_f32_bf16_T<<<(NKV * D + 255) / 256, 256, 0, stream>>>(vw, vwT, NKV, D);
  cvt_f32_bf16_T<<<(D * H * HD + 255) / 256, 256, 0, stream>>>(ow, owT, D, H * HD);

  // Projections; C tile = 64x128
  gemm_bf16_wmma<false><<<dim3(NQ / 128, M / 64), 128, 0, stream>>>(
      hs_bf, qwT, qg_bf, M, NQ, D);
  gemm_bf16_wmma<false><<<dim3(NKV / 128, M / 64), 128, 0, stream>>>(
      hs_bf, kwT, kr_bf, M, NKV, D);
  gemm_bf16_wmma<false><<<dim3(NKV / 128, M / 64), 128, 0, stream>>>(
      hs_bf, vwT, v_bf, M, NKV, D);

  // RMSNorm + RoPE: q row-major out; k d-major (transposed) out
  normrope<<<(M * H) / 4, 128, 0, stream>>>(qg_bf, q_bf, qnw, cs, sn, H, 2 * HD,
                                            M * H, 0, S);
  normrope<<<(M * KVH) / 4, 128, 0, stream>>>(kr_bf, kT_bf, knw, cs, sn, KVH, HD,
                                              M * KVH, 1, S);

  // Flash attention with fused sigmoid gating
  flash_attn<<<dim3(S / 16, H, B), 32, 0, stream>>>(q_bf, kT_bf, v_bf, qg_bf,
                                                    at_bf);

  // Output projection -> fp32 d_out
  gemm_bf16_wmma<true><<<dim3(D / 128, M / 64), 128, 0, stream>>>(
      at_bf, owT, d_out, M, D, H * HD);
}
